// GraphUNets_90486370992983
// MI455X (gfx1250) — compile-verified, matrix operand order checked by reading, not run on hardware
//
#include <hip/hip_runtime.h>
#include <math.h>

// ---------------- problem constants (from reference) ----------------
#define NN   8192
#define EE   262144
#define FIN  128
#define GG   64
#define CC   10
#define KK1  4096
#define KK2  2048
#define KK3  1024

typedef __attribute__((ext_vector_type(16))) _Float16 v16h;
typedef __attribute__((ext_vector_type(8)))  float    v8f;

__device__ __forceinline__ float eluf(float x) { return x > 0.f ? x : __expf(x) - 1.f; }

// =====================================================================
// WMMA GEMM: C[M,Nc] = A[M,K] @ B[K,Nc]; fp32 in/out, f16 multiplies.
// Each wave owns FOUR 16x16 C tiles stacked in M (64 rows x 16 cols),
// sharing one B fragment per k-step -> 4x v_wmma_f32_16x16x32_f16 per
// iteration and 1/4 the B traffic. 8 waves/block => 512 rows/block.
// All M here are multiples of 64, all K multiples of 32, Nc of 16.
// A-operand layout (ISA 7.12.2, 16-bit A 16x32): lane L holds row L&15;
// halves 0..7 -> k = base+0..7, halves 8..15 -> k = base+16..23, where
// base = 8 for lanes 16..31 (two contiguous float4 pairs per fragment).
// C/D layout: VGPR v holds M = v + 8*(L>=16), N = L&15.
// =====================================================================
__global__ void k_wmma_gemm(const float* __restrict__ A, const float* __restrict__ B,
                            float* __restrict__ C, int M, int K, int Nc) {
  const int wave = threadIdx.x >> 5;
  const int lane = threadIdx.x & 31;
  const int tm0  = (blockIdx.x * 8 + wave) * 4;   // first of 4 row-tiles
  const int tn   = blockIdx.y;
  if (tm0 * 16 >= M) return;                      // uniform per-wave exit
  const int r  = lane & 15;
  const int kg = (lane >> 4) << 3;                // +8 k offset for upper half-wave
  const float* Bcol = B + tn * 16 + r;
  const float* Ar0  = A + (size_t)(tm0 * 16 + r) * K;
  const size_t rowStride = (size_t)16 * K;        // one 16-row tile
  v8f acc0 = {}, acc1 = {}, acc2 = {}, acc3 = {};
  for (int k0 = 0; k0 < K; k0 += 32) {
    if (k0 + 32 < K)                              // prefetch next A k-slab
      __builtin_prefetch(Ar0 + k0 + 32 + kg, 0, 0);
    // ---- B fragment (shared by the 4 row tiles) ----
    v16h b;
#pragma unroll
    for (int h = 0; h < 16; ++h) {
      const int k = k0 + kg + h + ((h >> 3) << 3);
      b[h] = (_Float16)Bcol[(size_t)k * Nc];
    }
    // ---- 4 A fragments + 4 WMMAs ----
#pragma unroll
    for (int t = 0; t < 4; ++t) {
      const float* Ar = Ar0 + (size_t)t * rowStride + k0 + kg;
      const float4 q0 = *(const float4*)(Ar);
      const float4 q1 = *(const float4*)(Ar + 4);
      const float4 q2 = *(const float4*)(Ar + 16);
      const float4 q3 = *(const float4*)(Ar + 20);
      v16h a;
      a[0]  = (_Float16)q0.x; a[1]  = (_Float16)q0.y;
      a[2]  = (_Float16)q0.z; a[3]  = (_Float16)q0.w;
      a[4]  = (_Float16)q1.x; a[5]  = (_Float16)q1.y;
      a[6]  = (_Float16)q1.z; a[7]  = (_Float16)q1.w;
      a[8]  = (_Float16)q2.x; a[9]  = (_Float16)q2.y;
      a[10] = (_Float16)q2.z; a[11] = (_Float16)q2.w;
      a[12] = (_Float16)q3.x; a[13] = (_Float16)q3.y;
      a[14] = (_Float16)q3.z; a[15] = (_Float16)q3.w;
      if (t == 0)
        acc0 = __builtin_amdgcn_wmma_f32_16x16x32_f16(false, a, false, b,
                                                      (short)0, acc0, false, false);
      else if (t == 1)
        acc1 = __builtin_amdgcn_wmma_f32_16x16x32_f16(false, a, false, b,
                                                      (short)0, acc1, false, false);
      else if (t == 2)
        acc2 = __builtin_amdgcn_wmma_f32_16x16x32_f16(false, a, false, b,
                                                      (short)0, acc2, false, false);
      else
        acc3 = __builtin_amdgcn_wmma_f32_16x16x32_f16(false, a, false, b,
                                                      (short)0, acc3, false, false);
    }
  }
  const int colBase = tn * 16 + r;
  const int mB = (lane >> 4) << 3;
#pragma unroll
  for (int t = 0; t < 4; ++t) {
    const v8f acc = (t == 0) ? acc0 : (t == 1) ? acc1 : (t == 2) ? acc2 : acc3;
    const int rowBase = (tm0 + t) * 16 + mB;
#pragma unroll
    for (int v = 0; v < 8; ++v)
      C[(size_t)(rowBase + v) * Nc + colBase] = acc[v];
  }
}

// ---------------- elementwise ----------------
__global__ void k_elu(float* x, int n) {
  int i = blockIdx.x * 256 + threadIdx.x;
  if (i < n) x[i] = eluf(x[i]);
}
__global__ void k_fill_f(float* x, float v, int n) {
  int i = blockIdx.x * 256 + threadIdx.x;
  if (i < n) x[i] = v;
}
__global__ void k_zero_b(unsigned char* x, long n) {
  long i = (long)blockIdx.x * 256 + threadIdx.x;
  if (i < n) x[i] = 0;
}

// ---------------- sparse (level-0) GCN pieces ----------------
__global__ void k_edge_deg(const int* __restrict__ ei, float* deg) {
  int e = blockIdx.x * 256 + threadIdx.x;
  if (e < EE) atomicAdd(&deg[ei[EE + e]], 1.0f);       // column (dst) degree; dups accumulate
}
__global__ void k_dinv(const float* deg, float* dinv, int n) {
  int i = blockIdx.x * 256 + threadIdx.x;
  if (i < n) { float d = deg[i]; dinv[i] = d > 0.f ? rsqrtf(d) : 0.f; }
}
// out[d,f] = b[f] + dinv[d]^2 * t[d,f]   (the +I self-loop of A-hat)
__global__ void k_self_init(const float* __restrict__ t, const float* __restrict__ bias,
                            const float* __restrict__ dinv, float* out, int n, int F) {
  int i = blockIdx.x * 256 + threadIdx.x;
  if (i >= n * F) return;
  int d = i / F, f = i % F;
  float dv = dinv[d];
  out[i] = bias[f] + dv * dv * t[i];
}
// out[dst,f] += dinv[src]*dinv[dst]*t[src,f] per edge occurrence
__global__ void k_edge_agg(const int* __restrict__ ei, const float* __restrict__ t,
                           const float* __restrict__ dinv, float* out, int F) {
  long i = (long)blockIdx.x * 256 + threadIdx.x;
  if (i >= (long)EE * F) return;
  int e = (int)(i / F), f = (int)(i % F);
  int s = ei[e], d = ei[EE + e];
  atomicAdd(&out[(long)d * F + f], dinv[s] * dinv[d] * t[(long)s * F + f]);
}

// ---------------- dense (pooled-level) GCN pieces ----------------
__global__ void k_deg_dense(const unsigned char* __restrict__ A, float* deg, int n) {
  int d = blockIdx.x * 256 + threadIdx.x;
  if (d >= n) return;
  int c = 1;                                           // +1 self loop
  for (int s = 0; s < n; ++s) c += A[(long)s * n + d];
  deg[d] = (float)c;
}
// out[d,f] = b[f] + dinv[d]*( dinv[d]*t[d,f] + sum_s A[s,d]*dinv[s]*t[s,f] )
__global__ void k_agg_dense(const unsigned char* __restrict__ A, const float* __restrict__ t,
                            const float* __restrict__ dinv, const float* __restrict__ bias,
                            float* out, int n, int F) {
  const int d = blockIdx.x;            // one block per destination node
  const int tt = threadIdx.x;          // 128 threads over features
  float acc0 = 0.f, acc1 = 0.f;        // F <= 256
  for (int s = 0; s < n; ++s) {
    if (A[(long)s * n + d]) {
      float dv = dinv[s];
      acc0 += dv * t[(long)s * F + tt];
      if (F > 128) acc1 += dv * t[(long)s * F + tt + 128];
    }
  }
  float dd = dinv[d];
  out[(long)d * F + tt] = bias[tt] + dd * (dd * t[(long)d * F + tt] + acc0);
  if (F > 128)
    out[(long)d * F + tt + 128] = bias[tt + 128] + dd * (dd * t[(long)d * F + tt + 128] + acc1);
}

// ---------------- GPool: scores, rank-based top-k, gated gather ----------------
__global__ void k_scores(const float* __restrict__ x, const float* __restrict__ p,
                         float* s, int n, int F) {
  int i = blockIdx.x * 256 + threadIdx.x;
  if (i >= n) return;
  float nrm = 0.f;
  for (int f = 0; f < F; ++f) nrm += p[f] * p[f];
  nrm = sqrtf(nrm);
  float acc = 0.f;
  for (int f = 0; f < F; ++f) acc += x[(long)i * F + f] * p[f];
  s[i] = acc / nrm;
}
// rank[i] unique -> pooled position; reproduces top_k descending order
__global__ void k_topk_select(const float* __restrict__ s, int n, int K,
                              int* pos, int* idx, float* gate) {
  int i = blockIdx.x * 256 + threadIdx.x;
  if (i >= n) return;
  float si = s[i];
  int cnt = 0;
  for (int j = 0; j < n; ++j) {
    float sj = s[j];
    cnt += (sj > si) || (sj == si && j < i);
  }
  if (cnt < K) {
    pos[i] = cnt;
    idx[cnt] = i;
    gate[cnt] = 1.f / (1.f + __expf(-si));
  } else {
    pos[i] = -1;
  }
}
__global__ void k_pool_gather(const float* __restrict__ x, const int* __restrict__ idx,
                              const float* __restrict__ gate, float* xp, int K, int F) {
  int i = blockIdx.x * 256 + threadIdx.x;
  if (i >= K * F) return;
  int a = i / F, f = i % F;
  xp[i] = x[(long)idx[a] * F + f] * gate[a];
}

// ---------------- adjacency builds ----------------
// A2[pos[s],pos[d]] = 1 for surviving edges (idempotent byte store = rebinarize)
__global__ void k_adj_edges(const int* __restrict__ ei, const int* __restrict__ pos,
                            unsigned char* A2) {
  int e = blockIdx.x * 256 + threadIdx.x;
  if (e >= EE) return;
  int ps = pos[ei[e]], pd = pos[ei[EE + e]];
  if (ps >= 0 && pd >= 0) A2[(long)ps * KK1 + pd] = 1;
}
// Aout[a,b] = Ain[idx[a], idx[b]] (already 0/1)
__global__ void k_adj_sub(const unsigned char* __restrict__ Ain, int nin,
                          const int* __restrict__ idx, unsigned char* Aout, int nout) {
  long i = (long)blockIdx.x * 256 + threadIdx.x;
  if (i >= (long)nout * nout) return;
  int a = (int)(i / nout), b = (int)(i % nout);
  Aout[i] = Ain[(long)idx[a] * nin + idx[b]];
}

// ---------------- unpool + concat ----------------
__global__ void k_unpool_scatter(const float* __restrict__ xs, const int* __restrict__ idx,
                                 float* dst, int K, int Fs, int Fd) {
  int i = blockIdx.x * 256 + threadIdx.x;
  if (i >= K * Fs) return;
  int a = i / Fs, f = i % Fs;
  dst[(long)idx[a] * Fd + f] = xs[i];
}
__global__ void k_concat_fill(const float* __restrict__ xs, float* dst,
                              int n, int Fs, int Fd, int off) {
  int i = blockIdx.x * 256 + threadIdx.x;
  if (i >= n * Fs) return;
  int d = i / Fs, f = i % Fs;
  dst[(long)d * Fd + off + f] = xs[i];
}

// ---------------- readout: per-graph max & mean over sorted batch ----------------
__global__ void k_segpool(const float* __restrict__ x13, const int* __restrict__ batch,
                          float* h) {
  const int g = blockIdx.x;      // 64 graphs
  const int f = threadIdx.x;     // 32 feature channels (one wave)
  float mx = -INFINITY, sm = 0.f;
  int cnt = 0;
  for (int i = 0; i < NN; ++i) {
    if (batch[i] == g) {
      float v = x13[(long)i * 32 + f];
      mx = fmaxf(mx, v);
      sm += v;
      ++cnt;
    }
  }
  float mean = cnt > 0 ? sm / (float)cnt : 0.f;
  h[g * 64 + f]      = eluf(mx);     // h = elu(concat(gmax, gsum/cnt))
  h[g * 64 + 32 + f] = eluf(mean);
}

// ---------------- classifier head: h2 @ Wc + bc, log_softmax ----------------
__global__ void k_head(const float* __restrict__ h2, const float* __restrict__ Wc,
                       const float* __restrict__ bc, float* out) {
  int g = threadIdx.x;
  if (g >= GG) return;
  float l[CC];
  for (int c = 0; c < CC; ++c) {
    float acc = bc[c];
    for (int j = 0; j < 64; ++j) acc += h2[g * 64 + j] * Wc[j * CC + c];
    l[c] = acc;
  }
  float m = l[0];
  for (int c = 1; c < CC; ++c) m = fmaxf(m, l[c]);
  float lse = 0.f;
  for (int c = 0; c < CC; ++c) lse += __expf(l[c] - m);
  lse = __logf(lse);
  for (int c = 0; c < CC; ++c) out[g * CC + c] = l[c] - m - lse;
}

// ==================== host-side helpers ====================
static void gemm(const float* A, const float* B, float* C, int M, int K, int Nc,
                 hipStream_t s) {
  dim3 grid((unsigned)((M + 511) / 512), (unsigned)(Nc / 16));   // 512 rows per block
  k_wmma_gemm<<<grid, 256, 0, s>>>(A, B, C, M, K, Nc);
}
static void elu_ip(float* p, int n, hipStream_t s) {
  k_elu<<<(n + 255) / 256, 256, 0, s>>>(p, n);
}
static void gcn_sparse(const int* ei, const float* xin, const float* W, const float* bias,
                       int Fi, int Fo, float* t, float* deg, float* dinv, float* out,
                       hipStream_t s) {
  gemm(xin, W, t, NN, Fi, Fo, s);
  k_fill_f<<<(NN + 255) / 256, 256, 0, s>>>(deg, 1.0f, NN);
  k_edge_deg<<<(EE + 255) / 256, 256, 0, s>>>(ei, deg);
  k_dinv<<<(NN + 255) / 256, 256, 0, s>>>(deg, dinv, NN);
  k_self_init<<<(NN * Fo + 255) / 256, 256, 0, s>>>(t, bias, dinv, out, NN, Fo);
  long eag = (long)EE * Fo;
  k_edge_agg<<<(unsigned)((eag + 255) / 256), 256, 0, s>>>(ei, t, dinv, out, Fo);
  elu_ip(out, NN * Fo, s);
}
static void gcn_dense(const unsigned char* A, int n, const float* xin, const float* W,
                      const float* bias, int Fi, int Fo, float* t, float* deg, float* dinv,
                      float* out, hipStream_t s) {
  gemm(xin, W, t, n, Fi, Fo, s);
  k_deg_dense<<<(n + 255) / 256, 256, 0, s>>>(A, deg, n);
  k_dinv<<<(n + 255) / 256, 256, 0, s>>>(deg, dinv, n);
  k_agg_dense<<<n, 128, 0, s>>>(A, t, dinv, bias, out, n, Fo);
  elu_ip(out, n * Fo, s);
}
static void gpool(const float* xin, int n, int Fi, const float* p, int K,
                  float* sc, int* pos, int* idx, float* gate, float* xp, hipStream_t s) {
  k_scores<<<(n + 255) / 256, 256, 0, s>>>(xin, p, sc, n, Fi);
  k_topk_select<<<(n + 255) / 256, 256, 0, s>>>(sc, n, K, pos, idx, gate);
  k_pool_gather<<<(K * Fi + 255) / 256, 256, 0, s>>>(xin, idx, gate, xp, K, Fi);
  elu_ip(xp, K * Fi, s);
}
static void unpool_concat(const float* xs, const int* idx, int Ks, int Fs,
                          const float* xskip, int n, int Fskip, float* dst, hipStream_t s) {
  int Fd = Fs + Fskip;
  k_fill_f<<<(n * Fd + 255) / 256, 256, 0, s>>>(dst, 0.0f, n * Fd);
  k_unpool_scatter<<<(Ks * Fs + 255) / 256, 256, 0, s>>>(xs, idx, dst, Ks, Fs, Fd);
  k_concat_fill<<<(n * Fskip + 255) / 256, 256, 0, s>>>(xskip, dst, n, Fskip, Fd, Fs);
  elu_ip(dst, n * Fd, s);
}

extern "C" void kernel_launch(void* const* d_in, const int* in_sizes, int n_in,
                              void* d_out, int out_size, void* d_ws, size_t ws_size,
                              hipStream_t stream) {
  (void)in_sizes; (void)n_in; (void)out_size; (void)ws_size;
  const float* x    = (const float*)d_in[0];
  const int*   ei   = (const int*)d_in[1];     // [2,E] row-major: src=ei[e], dst=ei[E+e]
  const int*   bat  = (const int*)d_in[2];
  const float* W1 = (const float*)d_in[3];  const float* b1 = (const float*)d_in[4];
  const float* W2 = (const float*)d_in[5];  const float* b2 = (const float*)d_in[6];
  const float* W3 = (const float*)d_in[7];  const float* b3 = (const float*)d_in[8];
  const float* W4 = (const float*)d_in[9];  const float* b4 = (const float*)d_in[10];
  const float* W5 = (const float*)d_in[11]; const float* b5 = (const float*)d_in[12];
  const float* W6 = (const float*)d_in[13]; const float* b6 = (const float*)d_in[14];
  const float* W7 = (const float*)d_in[15]; const float* b7 = (const float*)d_in[16];
  const float* p1 = (const float*)d_in[17];
  const float* p2 = (const float*)d_in[18];
  const float* p3 = (const float*)d_in[19];
  const float* Wl1 = (const float*)d_in[20];
  const float* Wc  = (const float*)d_in[21]; const float* bc = (const float*)d_in[22];

  // ---- bump allocator over d_ws (~47 MB total) ----
  char* base = (char*)d_ws;
  size_t off = 0;
  auto alloc = [&](size_t bytes) -> void* {
    void* p = base + off;
    off = (off + bytes + 255) & ~(size_t)255;
    return p;
  };
  float* deg0  = (float*)alloc(NN * 4);        float* dinv0 = (float*)alloc(NN * 4);
  float* t0    = (float*)alloc((size_t)NN * 32 * 4);
  float* x1    = (float*)alloc((size_t)NN * 32 * 4);
  float* s1    = (float*)alloc(NN * 4);
  int*   pos1  = (int*)alloc(NN * 4);
  int*   idx1  = (int*)alloc(KK1 * 4);         float* gate1 = (float*)alloc(KK1 * 4);
  float* x2    = (float*)alloc((size_t)KK1 * 32 * 4);
  unsigned char* A2 = (unsigned char*)alloc((size_t)KK1 * KK1);
  float* deg2  = (float*)alloc(KK1 * 4);       float* dinv2 = (float*)alloc(KK1 * 4);
  float* t2    = (float*)alloc((size_t)KK1 * 64 * 4);
  float* x3    = (float*)alloc((size_t)KK1 * 64 * 4);
  float* s2    = (float*)alloc(KK1 * 4);
  int*   pos2  = (int*)alloc(KK1 * 4);
  int*   idx2  = (int*)alloc(KK2 * 4);         float* gate2 = (float*)alloc(KK2 * 4);
  float* x4    = (float*)alloc((size_t)KK2 * 64 * 4);
  unsigned char* A4 = (unsigned char*)alloc((size_t)KK2 * KK2);
  float* deg4  = (float*)alloc(KK2 * 4);       float* dinv4 = (float*)alloc(KK2 * 4);
  float* t4    = (float*)alloc((size_t)KK2 * 128 * 4);
  float* x5    = (float*)alloc((size_t)KK2 * 128 * 4);
  float* s3    = (float*)alloc(KK2 * 4);
  int*   pos3  = (int*)alloc(KK2 * 4);
  int*   idx3  = (int*)alloc(KK3 * 4);         float* gate3 = (float*)alloc(KK3 * 4);
  float* x6    = (float*)alloc((size_t)KK3 * 128 * 4);
  unsigned char* A6 = (unsigned char*)alloc((size_t)KK3 * KK3);
  float* deg6  = (float*)alloc(KK3 * 4);       float* dinv6 = (float*)alloc(KK3 * 4);
  float* t6    = (float*)alloc((size_t)KK3 * 256 * 4);
  float* x7    = (float*)alloc((size_t)KK3 * 256 * 4);
  float* x8    = (float*)alloc((size_t)KK2 * 384 * 4);
  float* t8    = (float*)alloc((size_t)KK2 * 128 * 4);
  float* x9    = (float*)alloc((size_t)KK2 * 128 * 4);
  float* x10   = (float*)alloc((size_t)KK1 * 192 * 4);
  float* t10   = (float*)alloc((size_t)KK1 * 64 * 4);
  float* x11   = (float*)alloc((size_t)KK1 * 64 * 4);
  float* x12   = (float*)alloc((size_t)NN * 96 * 4);
  float* t12   = (float*)alloc((size_t)NN * 32 * 4);
  float* x13   = (float*)alloc((size_t)NN * 32 * 4);
  float* h     = (float*)alloc(GG * 64 * 4);
  float* h2    = (float*)alloc(GG * 64 * 4);

  // ---- encoder ----
  gcn_sparse(ei, x, W1, b1, FIN, 32, t0, deg0, dinv0, x1, stream);              // x1
  gpool(x1, NN, 32, p1, KK1, s1, pos1, idx1, gate1, x2, stream);                // x2, i2
  k_zero_b<<<(unsigned)(((long)KK1 * KK1 + 255) / 256), 256, 0, stream>>>(A2, (long)KK1 * KK1);
  k_adj_edges<<<(EE + 255) / 256, 256, 0, stream>>>(ei, pos1, A2);              // A2
  gcn_dense(A2, KK1, x2, W2, b2, 32, 64, t2, deg2, dinv2, x3, stream);          // x3
  gpool(x3, KK1, 64, p2, KK2, s2, pos2, idx2, gate2, x4, stream);               // x4, i4
  k_adj_sub<<<(unsigned)(((long)KK2 * KK2 + 255) / 256), 256, 0, stream>>>(A2, KK1, idx2, A4, KK2);
  gcn_dense(A4, KK2, x4, W3, b3, 64, 128, t4, deg4, dinv4, x5, stream);         // x5
  gpool(x5, KK2, 128, p3, KK3, s3, pos3, idx3, gate3, x6, stream);              // x6, i6
  k_adj_sub<<<(unsigned)(((long)KK3 * KK3 + 255) / 256), 256, 0, stream>>>(A4, KK2, idx3, A6, KK3);
  gcn_dense(A6, KK3, x6, W4, b4, 128, 256, t6, deg6, dinv6, x7, stream);        // x7

  // ---- decoder ----
  unpool_concat(x7, idx3, KK3, 256, x5, KK2, 128, x8, stream);                  // x8 [K2,384]
  gcn_dense(A4, KK2, x8, W5, b5, 384, 128, t8, deg4, dinv4, x9, stream);        // x9
  unpool_concat(x9, idx2, KK2, 128, x3, KK1, 64, x10, stream);                  // x10 [K1,192]
  gcn_dense(A2, KK1, x10, W6, b6, 192, 64, t10, deg2, dinv2, x11, stream);      // x11
  unpool_concat(x11, idx1, KK1, 64, x1, NN, 32, x12, stream);                   // x12 [N,96]
  gcn_sparse(ei, x12, W7, b7, 96, 32, t12, deg0, dinv0, x13, stream);           // x13

  // ---- readout + head ----
  k_segpool<<<GG, 32, 0, stream>>>(x13, bat, h);                                // h [G,64]
  gemm(h, Wl1, h2, GG, 64, 64, stream);                                         // h @ Wl1 (WMMA)
  elu_ip(h2, GG * 64, stream);
  k_head<<<1, 64, 0, stream>>>(h2, Wc, bc, (float*)d_out);                      // log_softmax
}
